// WSWGAT_3186865734210
// MI455X (gfx1250) — matrix operand order repaired and checked by reading, not compiled, not run on hardware
//
#include <hip/hip_runtime.h>
#include <hip/hip_bf16.h>

#define NN 50000
#define EE 800000
#define DD 128
#define HH 8
#define OO 16
#define FFN 512
#define LN_EPS 1e-6f
#define LRELU 0.01f

typedef _Float16 half_t;
typedef __attribute__((ext_vector_type(16))) _Float16 v16h;
typedef __attribute__((ext_vector_type(8)))  _Float16 v8h;
typedef __attribute__((ext_vector_type(4)))  _Float16 v4h;
typedef __attribute__((ext_vector_type(8)))  float    v8f;

// ---------------- WMMA fragment loaders (CDNA5 16x16x32 f16 layouts) ----------------
// A: 16x32 (MxK) row-major tile, ld in halves.
// lane: m = lane&15, g = lane>>4. halves j=0..15 -> k = g*16+j
__device__ __forceinline__ v16h load_a_frag(const half_t* __restrict__ tile, int ld, int ktile, int lane) {
    int m = lane & 15;
    int g = lane >> 4;
    const half_t* p = tile + (size_t)m * ld + ktile + g * 16;
    v8h lo = *(const v8h*)(p);
    v8h hi = *(const v8h*)(p + 8);
    return __builtin_shufflevector(lo, hi, 0,1,2,3,4,5,6,7,8,9,10,11,12,13,14,15);
}
// B: 32x16 (KxN). Stored N-major (bt[n][k], row length ldk halves).
// lane: n = lane&15, halves j=0..15 -> k = (lane>>4)*16 + j  => one contiguous 32B load.
__device__ __forceinline__ v16h load_b_frag(const half_t* __restrict__ bt, int ldk, int n0, int ktile, int lane) {
    int n = lane & 15;
    int g = lane >> 4;
    const half_t* p = bt + (size_t)(n0 + n) * ldk + ktile + g * 16;
    return *(const v16h*)(p);
}

__device__ __forceinline__ float lrelu_f(float v) { return v > 0.f ? v : LRELU * v; }

// ---------------- K0: repack weights to fp16 (N-major), convert x, zero counters ----------------
__global__ void k_prep(const float* __restrict__ x, const float* __restrict__ fcw,
                       const float* __restrict__ w1, const float* __restrict__ w2,
                       half_t* __restrict__ x16, half_t* __restrict__ wfct,
                       half_t* __restrict__ w1t, half_t* __restrict__ w2t,
                       int* __restrict__ cnt, int* __restrict__ cur) {
    int stride = gridDim.x * blockDim.x;
    for (int i = blockIdx.x * blockDim.x + threadIdx.x; i < NN * DD / 4; i += stride) {
        float4 v = *(const float4*)(x + (size_t)i * 4);
        v4h h;
        h[0] = (half_t)v.x; h[1] = (half_t)v.y; h[2] = (half_t)v.z; h[3] = (half_t)v.w;
        *(v4h*)(x16 + (size_t)i * 4) = h;
        if (i < DD * DD) {             // wfct[c][d] = fc_w[h][d][o], c = h*16+o
            int c = i >> 7, d = i & 127;
            int hh = c >> 4, o = c & 15;
            wfct[i] = (half_t)fcw[(hh * DD + d) * OO + o];
        }
        if (i < FFN * DD) {            // w1t[f][d] = w1[d][f] ; w2t[dc][f] = w2[f][dc]
            int f = i >> 7, d = i & 127;
            w1t[i] = (half_t)w1[d * FFN + f];
            int dc = i >> 9, f2 = i & 511;
            w2t[i] = (half_t)w2[f2 * DD + dc];
        }
        if (i < NN) { cnt[i] = 0; cur[i] = 0; }
    }
}

// ---------------- K1: z = x @ Wfc (WMMA), fused el/er ----------------
__global__ __launch_bounds__(256) void k_proj(const half_t* __restrict__ x16,
                                              const half_t* __restrict__ wfct,
                                              const float* __restrict__ a_l,
                                              const float* __restrict__ a_r,
                                              float* __restrict__ z,
                                              float* __restrict__ el,
                                              float* __restrict__ er) {
    int lane = threadIdx.x & 31;
    int w = threadIdx.x >> 5;                 // column tile == head (0..7)
    int row0 = blockIdx.x * 16;
    const half_t* atile = x16 + (size_t)row0 * DD;
    v8f c = {};
#pragma unroll
    for (int kt = 0; kt < DD; kt += 32) {
        v16h a = load_a_frag(atile, DD, kt, lane);
        v16h b = load_b_frag(wfct, DD, w * 16, kt, lane);
        c = __builtin_amdgcn_wmma_f32_16x16x32_f16(false, a, false, b, (short)0, c, false, false);
    }
    int n = lane & 15, g = lane >> 4;
    float aln = a_l[w * 16 + n];
    float arn = a_r[w * 16 + n];
#pragma unroll
    for (int r = 0; r < 8; ++r) {
        int m = row0 + g * 8 + r;
        z[(size_t)m * DD + w * 16 + n] = c[r];
        float pl = c[r] * aln;
        float pr = c[r] * arn;
#pragma unroll
        for (int s = 8; s >= 1; s >>= 1) { pl += __shfl_xor(pl, s); pr += __shfl_xor(pr, s); }
        if (n == 0) { el[m * HH + w] = pl; er[m * HH + w] = pr; }
    }
}

// ---------------- K2a/b/c: CSR build (count, scan, fill) ----------------
__global__ void k_count(const int* __restrict__ dst, int* __restrict__ cnt) {
    int stride = gridDim.x * blockDim.x;
    for (int e = blockIdx.x * blockDim.x + threadIdx.x; e < EE; e += stride)
        atomicAdd(&cnt[dst[e]], 1);
}

__global__ __launch_bounds__(1024) void k_scan(const int* __restrict__ cnt, int* __restrict__ off) {
    __shared__ int s[1024];
    __shared__ int carry_s;
    int t = threadIdx.x;
    if (t == 0) { off[0] = 0; carry_s = 0; }
    __syncthreads();
    for (int base = 0; base < NN; base += 1024) {
        s[t] = (base + t < NN) ? cnt[base + t] : 0;
        __syncthreads();
        for (int d = 1; d < 1024; d <<= 1) {
            int add = (t >= d) ? s[t - d] : 0;
            __syncthreads();
            s[t] += add;
            __syncthreads();
        }
        int carry = carry_s;
        __syncthreads();
        if (base + t < NN) off[base + t + 1] = s[t] + carry;
        if (t == 0) carry_s = carry + s[1023];
        __syncthreads();
    }
}

__global__ void k_fill(const int* __restrict__ src, const int* __restrict__ dst,
                       const int* __restrict__ off, int* __restrict__ cur,
                       int* __restrict__ csrc) {
    int stride = gridDim.x * blockDim.x;
    for (int e = blockIdx.x * blockDim.x + threadIdx.x; e < EE; e += stride) {
        int d = dst[e];
        int p = atomicAdd(&cur[d], 1);
        csrc[off[d] + p] = src[e];
    }
}

// ---------------- K3: per-node edge softmax + aggregation + elu + residual + LayerNorm ----------------
__global__ __launch_bounds__(256) void k_gat_agg(const int* __restrict__ off,
                                                 const int* __restrict__ csrc,
                                                 const float* __restrict__ el,
                                                 const float* __restrict__ er,
                                                 const float* __restrict__ z,
                                                 const float* __restrict__ x,
                                                 const float* __restrict__ ln_g,
                                                 const float* __restrict__ ln_b,
                                                 float* __restrict__ h1,
                                                 half_t* __restrict__ ln16) {
    int lane = threadIdx.x & 31;
    int node = blockIdx.x * 8 + (threadIdx.x >> 5);   // grid sized exactly: 6250*8 = 50000
    int beg = off[node], end = off[node + 1];
    int deg = end - beg;

    v8f ern = *(const v8f*)(er + (size_t)node * HH);  // 32B contiguous
    float mx[HH], sm[HH];
#pragma unroll
    for (int h = 0; h < HH; ++h) { mx[h] = -1e30f; sm[h] = 0.f; }

    // pass 1: per-head max over incoming edges (vector load of el row)
    for (int i = lane; i < deg; i += 32) {
        int s = csrc[beg + i];
        v8f elv = *(const v8f*)(el + (size_t)s * HH);
#pragma unroll
        for (int h = 0; h < HH; ++h)
            mx[h] = fmaxf(mx[h], lrelu_f(elv[h] + ern[h]));
    }
#pragma unroll
    for (int h = 0; h < HH; ++h)
#pragma unroll
        for (int s = 16; s >= 1; s >>= 1) mx[h] = fmaxf(mx[h], __shfl_xor(mx[h], s));

    // pass 2: per-head exp-sum
    for (int i = lane; i < deg; i += 32) {
        int s = csrc[beg + i];
        v8f elv = *(const v8f*)(el + (size_t)s * HH);
#pragma unroll
        for (int h = 0; h < HH; ++h)
            sm[h] += expf(lrelu_f(elv[h] + ern[h]) - mx[h]);
    }
#pragma unroll
    for (int h = 0; h < HH; ++h)
#pragma unroll
        for (int s = 16; s >= 1; s >>= 1) sm[h] += __shfl_xor(sm[h], s);

    // pass 3: lane owns feature slice [4*lane, 4*lane+4), head h_my = lane>>2
    int h_my = lane >> 2;
    float m_my = mx[0], s_my = sm[0], e_my = ern[0];
#pragma unroll
    for (int h = 1; h < HH; ++h)
        if (h_my == h) { m_my = mx[h]; s_my = sm[h]; e_my = ern[h]; }

    int c0 = lane * 4;
    float4 acc = make_float4(0.f, 0.f, 0.f, 0.f);
    for (int i = 0; i < deg; ++i) {
        int s = csrc[beg + i];
        float ev = lrelu_f(el[s * HH + h_my] + e_my);
        float al = expf(ev - m_my) / s_my;
        float4 zz = *(const float4*)(z + (size_t)s * DD + c0);
        acc.x += al * zz.x; acc.y += al * zz.y; acc.z += al * zz.z; acc.w += al * zz.w;
    }
    float4 xr = *(const float4*)(x + (size_t)node * DD + c0);
    float4 o;
    o.x = (acc.x > 0.f ? acc.x : expm1f(acc.x)) + xr.x;
    o.y = (acc.y > 0.f ? acc.y : expm1f(acc.y)) + xr.y;
    o.z = (acc.z > 0.f ? acc.z : expm1f(acc.z)) + xr.z;
    o.w = (acc.w > 0.f ? acc.w : expm1f(acc.w)) + xr.w;
    *(float4*)(h1 + (size_t)node * DD + c0) = o;

    // fused LayerNorm over the 128-wide row this wave owns -> fp16 for the FFN GEMM
    float s1 = o.x + o.y + o.z + o.w;
    float s2 = o.x * o.x + o.y * o.y + o.z * o.z + o.w * o.w;
#pragma unroll
    for (int s = 16; s >= 1; s >>= 1) { s1 += __shfl_xor(s1, s); s2 += __shfl_xor(s2, s); }
    float mu = s1 * (1.f / DD);
    float var = s2 * (1.f / DD) - mu * mu;
    float rs = rsqrtf(var + LN_EPS);
    float4 gg = *(const float4*)(ln_g + c0);
    float4 bb = *(const float4*)(ln_b + c0);
    v4h lo;
    lo[0] = (half_t)((o.x - mu) * rs * gg.x + bb.x);
    lo[1] = (half_t)((o.y - mu) * rs * gg.y + bb.y);
    lo[2] = (half_t)((o.z - mu) * rs * gg.z + bb.z);
    lo[3] = (half_t)((o.w - mu) * rs * gg.w + bb.w);
    *(v4h*)(ln16 + (size_t)node * DD + c0) = lo;
}

// ---------------- K5: fused FFN: out = gelu(ln@w1+b1)@w2 + b2 + h1 (double WMMA) ----------------
__global__ __launch_bounds__(256) void k_ffn(const half_t* __restrict__ ln16,
                                             const half_t* __restrict__ w1t,
                                             const half_t* __restrict__ w2t,
                                             const float* __restrict__ b1,
                                             const float* __restrict__ b2,
                                             const float* __restrict__ h1,
                                             float* __restrict__ out) {
    __shared__ half_t inter[16 * FFN];        // 16KB
    int lane = threadIdx.x & 31;
    int w = threadIdx.x >> 5;                 // 0..7
    int row0 = blockIdx.x * 16;
    int n = lane & 15, g = lane >> 4;
    const half_t* atile = ln16 + (size_t)row0 * DD;

    // hoist A fragments (shared across the 4 column tiles of pass 1)
    v16h af[4];
#pragma unroll
    for (int ki = 0; ki < 4; ++ki) af[ki] = load_a_frag(atile, DD, ki * 32, lane);

    // pass 1: inter[16 x 512], wave w owns columns [w*64, w*64+64)
#pragma unroll
    for (int ct = 0; ct < 4; ++ct) {
        int col0 = w * 64 + ct * 16;
        v8f c = {};
#pragma unroll
        for (int ki = 0; ki < 4; ++ki) {
            v16h bfr = load_b_frag(w1t, DD, col0, ki * 32, lane);
            c = __builtin_amdgcn_wmma_f32_16x16x32_f16(false, af[ki], false, bfr, (short)0, c, false, false);
        }
        float bias = b1[col0 + n];
#pragma unroll
        for (int r = 0; r < 8; ++r) {
            float v = c[r] + bias;
            v = 0.5f * v * (1.f + erff(v * 0.70710678118654752f));   // exact gelu
            inter[(g * 8 + r) * FFN + col0 + n] = (half_t)v;
        }
    }
    __syncthreads();

    // pass 2: out[16 x 128], wave w owns columns [w*16, w*16+16), K = 512
    int col0 = w * 16;
    v8f c = {};
#pragma unroll
    for (int kt = 0; kt < FFN; kt += 32) {
        v16h a = load_a_frag(inter, FFN, kt, lane);
        v16h bfr = load_b_frag(w2t, FFN, col0, kt, lane);
        c = __builtin_amdgcn_wmma_f32_16x16x32_f16(false, a, false, bfr, (short)0, c, false, false);
    }
    float bias = b2[col0 + n];
#pragma unroll
    for (int r = 0; r < 8; ++r) {
        int m = row0 + g * 8 + r;
        size_t idx = (size_t)m * DD + col0 + n;
        out[idx] = c[r] + bias + h1[idx];
    }
}

// ---------------- host ----------------
extern "C" void kernel_launch(void* const* d_in, const int* in_sizes, int n_in,
                              void* d_out, int out_size, void* d_ws, size_t ws_size,
                              hipStream_t stream) {
    (void)in_sizes; (void)n_in; (void)out_size; (void)ws_size;
    const float* x    = (const float*)d_in[0];
    const float* fcw  = (const float*)d_in[1];
    const float* a_l  = (const float*)d_in[2];
    const float* a_r  = (const float*)d_in[3];
    const float* ln_g = (const float*)d_in[4];
    const float* ln_b = (const float*)d_in[5];
    const float* w1   = (const float*)d_in[6];
    const float* b1   = (const float*)d_in[7];
    const float* w2   = (const float*)d_in[8];
    const float* b2   = (const float*)d_in[9];
    const int*   src  = (const int*)d_in[10];
    const int*   dst  = (const int*)d_in[11];
    float* out = (float*)d_out;

    char* ws = (char*)d_ws;
    size_t o = 0;
    auto alloc = [&](size_t bytes) -> void* {
        void* p = ws + o;
        o = (o + bytes + 255) & ~(size_t)255;
        return p;
    };
    half_t* x16  = (half_t*)alloc((size_t)NN * DD * 2);
    half_t* wfct = (half_t*)alloc((size_t)DD * DD * 2);
    half_t* w1t  = (half_t*)alloc((size_t)FFN * DD * 2);
    half_t* w2t  = (half_t*)alloc((size_t)DD * FFN * 2);
    float*  z    = (float*)alloc((size_t)NN * DD * 4);
    float*  el   = (float*)alloc((size_t)NN * HH * 4);
    float*  er   = (float*)alloc((size_t)NN * HH * 4);
    int*    cnt  = (int*)alloc((size_t)NN * 4);
    int*    off  = (int*)alloc((size_t)(NN + 1) * 4);
    int*    cur  = (int*)alloc((size_t)NN * 4);
    int*    csrc = (int*)alloc((size_t)EE * 4);
    float*  h1   = (float*)alloc((size_t)NN * DD * 4);
    half_t* ln16 = (half_t*)alloc((size_t)NN * DD * 2);

    k_prep<<<4096, 256, 0, stream>>>(x, fcw, w1, w2, x16, wfct, w1t, w2t, cnt, cur);
    k_count<<<3125, 256, 0, stream>>>(dst, cnt);
    k_scan<<<1, 1024, 0, stream>>>(cnt, off);
    k_fill<<<3125, 256, 0, stream>>>(src, dst, off, cur, csrc);
    k_proj<<<3125, 256, 0, stream>>>(x16, wfct, a_l, a_r, z, el, er);                 // 50000/16 row tiles
    k_gat_agg<<<6250, 256, 0, stream>>>(off, csrc, el, er, z, x, ln_g, ln_b, h1, ln16); // 1 wave / node, fused LN
    k_ffn<<<3125, 256, 0, stream>>>(ln16, w1t, w2t, b1, b2, h1, out);                 // 16-row tiles
}
